// GIBGCN_25872882991415
// MI455X (gfx1250) — compile-verified, matrix operand order checked by reading, not run on hardware
//
#include <hip/hip_runtime.h>
#include <hip/hip_bf16.h>
#include <math.h>

// ---------------- problem constants (match reference) ----------------
#define N_NODES   51200
#define N_EDGES   819200
#define N_GRAPHS  512
#define F_IN      128
#define HID       512
#define N_CLS     6
#define EPS_F     1e-5f

typedef __attribute__((ext_vector_type(16))) __bf16 v16bf;
typedef __attribute__((ext_vector_type(8)))  __bf16 v8bf;
typedef __attribute__((ext_vector_type(4)))  __bf16 v4bf;
typedef __attribute__((ext_vector_type(8)))  float  v8f;
typedef __attribute__((ext_vector_type(4)))  float  v4f;

// ---------------- tiny utility kernels ----------------
__global__ void zero_f32_kernel(float* p, int n) {
    int i = blockIdx.x * blockDim.x + threadIdx.x;
    if (i < n) p[i] = 0.0f;
}

__global__ void deg_kernel(const int* __restrict__ dst, float* __restrict__ deg, int E) {
    int e = blockIdx.x * blockDim.x + threadIdx.x;
    if (e < E) atomicAdd(&deg[dst[e]], 1.0f);
}

__global__ void dis_kernel(const float* __restrict__ deg, float* __restrict__ dis, int N) {
    int i = blockIdx.x * blockDim.x + threadIdx.x;
    if (i < N) dis[i] = rsqrtf(deg[i] + 1.0f);
}

__global__ void enorm_kernel(const int* __restrict__ src, const int* __restrict__ dst,
                             const float* __restrict__ dis, float* __restrict__ enorm, int E) {
    int e = blockIdx.x * blockDim.x + threadIdx.x;
    if (e < E) enorm[e] = dis[src[e]] * dis[dst[e]];
}

// transpose + convert weight [K,FO] f32 -> [FO,K] bf16 (contiguous-K fragment loads)
__global__ void wt_kernel(const float* __restrict__ W, __bf16* __restrict__ Wt, int K, int FO) {
    int i = blockIdx.x * blockDim.x + threadIdx.x;
    if (i >= K * FO) return;
    int fo = i / K, k = i % K;
    Wt[(size_t)fo * K + k] = (__bf16)W[(size_t)k * FO + fo];
}

// ---------------- WMMA bf16 GEMM: Out[M,FO] = A[M,K] @ Wt[FO,K]^T ----------------
// MODE 0: OutB = bf16(val)                       (no bias)
// MODE 1: OutB = bf16(tanh(val + bias))
// MODE 2: OutF = relu(val + bias)
// Ping-pong software pipeline (unroll-by-2): compute frag0 while loading frag1
// and vice versa -> no rotate copies, and in-flight loads never write the
// WMMA's live A/B registers (avoids WMMA->VALU WAR hazard NOPs).
// K/32 is even (4 or 16) at every call site.
template <int MODE>
__global__ __launch_bounds__(256) void gemm_bf16_kernel(
    const float* __restrict__ A, const __bf16* __restrict__ Wt,
    const float* __restrict__ bias, float* __restrict__ OutF,
    __bf16* __restrict__ OutB, int M, int K, int FO)
{
    const int lane = threadIdx.x & 31;
    const int wave = threadIdx.x >> 5;
    const int tilesCol = FO >> 6;                 // 64 cols per wave-tile
    const int tile = blockIdx.x * 8 + wave;
    const int totalTiles = (M >> 4) * tilesCol;
    if (tile >= totalTiles) return;               // wave-uniform: EXEC stays all-ones
    const int rowTile = tile / tilesCol;
    const int colTile = tile % tilesCol;
    const int half = lane >> 4;
    const int l15  = lane & 15;
    const int m = rowTile * 16 + l15;
    const float* arow = A + (size_t)m * K;

    const __bf16* brow[4];
    #pragma unroll
    for (int j = 0; j < 4; ++j)
        brow[j] = Wt + (size_t)(colTile * 64 + j * 16 + l15) * K;

    v8f acc[4] = {};

    struct Frag { v4f a0, a1, a2, a3; v8bf blo[4], bhi[4]; };
    Frag f0, f1;

    auto loadFrag = [&](Frag& f, int kb) {
        const int k0 = kb + half * 8;
        const int k1 = k0 + 16;
        f.a0 = *(const v4f*)(arow + k0);
        f.a1 = *(const v4f*)(arow + k0 + 4);
        f.a2 = *(const v4f*)(arow + k1);
        f.a3 = *(const v4f*)(arow + k1 + 4);
        #pragma unroll
        for (int j = 0; j < 4; ++j) {
            f.blo[j] = *(const v8bf*)(brow[j] + k0);
            f.bhi[j] = *(const v8bf*)(brow[j] + k1);
        }
    };
    auto computeFrag = [&](const Frag& f) {
        v16bf af;
        #pragma unroll
        for (int i = 0; i < 4; ++i) {
            af[i]      = (__bf16)f.a0[i];
            af[4 + i]  = (__bf16)f.a1[i];
            af[8 + i]  = (__bf16)f.a2[i];
            af[12 + i] = (__bf16)f.a3[i];
        }
        #pragma unroll
        for (int j = 0; j < 4; ++j) {
            v16bf bfv;
            #pragma unroll
            for (int i = 0; i < 8; ++i) { bfv[i] = f.blo[j][i]; bfv[8 + i] = f.bhi[j][i]; }
            acc[j] = __builtin_amdgcn_wmma_f32_16x16x32_bf16(
                false, af, false, bfv, (short)0, acc[j], false, false);
        }
    };

    const int S = K >> 5;          // number of 32-wide K-steps (even: 4 or 16)
    loadFrag(f0, 0);
    for (int s = 0; s < S - 2; s += 2) {
        loadFrag(f1, (s + 1) * 32);   // overlaps the 4 WMMAs of f0
        computeFrag(f0);
        loadFrag(f0, (s + 2) * 32);   // overlaps the 4 WMMAs of f1
        computeFrag(f1);
    }
    // tail pair (S-2, S-1)
    loadFrag(f1, (S - 1) * 32);
    computeFrag(f0);
    computeFrag(f1);

    // epilogue: C/D layout -> row = rowTile*16 + half*8 + r, col = colTile*64 + j*16 + l15
    #pragma unroll
    for (int j = 0; j < 4; ++j) {
        const int n = colTile * 64 + j * 16 + l15;
        const float bn = (MODE != 0) ? bias[n] : 0.0f;   // one load per column tile
        #pragma unroll
        for (int r = 0; r < 8; ++r) {
            const int mr = rowTile * 16 + half * 8 + r;
            const size_t idx = (size_t)mr * FO + n;
            float v = acc[j][r];
            if (MODE == 0) {
                OutB[idx] = (__bf16)v;
            } else if (MODE == 1) {
                OutB[idx] = (__bf16)tanhf(v + bn);
            } else {
                OutF[idx] = fmaxf(v + bn, 0.0f);
            }
        }
    }
}

// ---------------- GCN aggregation ----------------
// out[n,:] = dis[n]^2 * t[n,:] + b  (self-loop + bias), vectorized 4-wide
__global__ void agg_init_kernel(const __bf16* __restrict__ t, const float* __restrict__ dis,
                                const float* __restrict__ bias, float* __restrict__ out, int N) {
    long long id = (long long)blockIdx.x * blockDim.x + threadIdx.x;
    if (id >= (long long)N * (HID / 4)) return;
    int n = (int)(id >> 7);          // HID/4 = 128
    int c = (int)(id & 127);
    float d2 = dis[n] * dis[n];
    v4bf t4 = *(const v4bf*)(t + (size_t)n * HID + 4 * c);
    v4f  b4 = *(const v4f*)(bias + 4 * c);
    v4f  o4;
    #pragma unroll
    for (int i = 0; i < 4; ++i) o4[i] = d2 * (float)t4[i] + b4[i];
    *(v4f*)(out + (size_t)n * HID + 4 * c) = o4;
}

// out[dst,:] += enorm[e] * t[src,:]  (scatter-add, 128 lanes per edge)
__global__ void agg_scatter_kernel(const __bf16* __restrict__ t, const int* __restrict__ src,
                                   const int* __restrict__ dst, const float* __restrict__ enorm,
                                   float* __restrict__ out, int E) {
    long long id = (long long)blockIdx.x * blockDim.x + threadIdx.x;
    if (id >= (long long)E * (HID / 4)) return;
    int e = (int)(id >> 7);
    int c = (int)(id & 127);
    float en = enorm[e];
    int s = src[e], d = dst[e];
    v4bf t4 = *(const v4bf*)(t + (size_t)s * HID + 4 * c);
    float* op = out + (size_t)d * HID + 4 * c;
    #pragma unroll
    for (int i = 0; i < 4; ++i) atomicAdd(op + i, en * (float)t4[i]);
}

// ---------------- cluster assignment: softmax(U @ Wc2 + bc2), U = tanh(...) bf16 ----------------
__global__ void assign_kernel(const __bf16* __restrict__ U, const float* __restrict__ Wc2,
                              const float* __restrict__ bc2, float* __restrict__ assign, int N) {
    int n = blockIdx.x * blockDim.x + threadIdx.x;
    if (n >= N) return;
    float s0 = bc2[0], s1 = bc2[1];
    const __bf16* u = U + (size_t)n * HID;
    for (int k = 0; k < HID; ++k) {
        float uv = (float)u[k];
        s0 += uv * Wc2[2 * k];
        s1 += uv * Wc2[2 * k + 1];
    }
    float mx = fmaxf(s0, s1);
    float e0 = expf(s0 - mx), e1 = expf(s1 - mx);
    float inv = 1.0f / (e0 + e1);
    assign[2 * n]     = e0 * inv;
    assign[2 * n + 1] = e1 * inv;
}

// pos[g,:] += assign[n,0] * h[n,:]
__global__ void pos_kernel(const float* __restrict__ h, const float* __restrict__ assign,
                           const int* __restrict__ batch, float* __restrict__ posraw, int N) {
    long long id = (long long)blockIdx.x * blockDim.x + threadIdx.x;
    if (id >= (long long)N * (HID / 4)) return;
    int n = (int)(id >> 7);
    int c = (int)(id & 127);
    float a0 = assign[2 * n];
    int g = batch[n];
    v4f h4 = *(const v4f*)(h + (size_t)n * HID + 4 * c);
    float* pp = posraw + (size_t)g * HID + 4 * c;
    #pragma unroll
    for (int i = 0; i < 4; ++i) atomicAdd(pp + i, a0 * h4[i]);
}

// column sums of h (for gemb = mean over all nodes): grid = row-blocks, block = 512 cols
__global__ void colsum_kernel(const float* __restrict__ h, float* __restrict__ colsum, int N) {
    int f = threadIdx.x;               // 512 threads
    int rowsPer = N / gridDim.x;
    int r0 = blockIdx.x * rowsPer;
    float s = 0.0f;
    for (int n = r0; n < r0 + rowsPer; ++n) s += h[(size_t)n * HID + f];
    atomicAdd(&colsum[f], s);
}

// new_adj[g] += outer(assign[src], assign[dst])
__global__ void newadj_kernel(const float* __restrict__ assign, const int* __restrict__ src,
                              const int* __restrict__ dst, const int* __restrict__ batch,
                              float* __restrict__ newadj, int E) {
    int e = blockIdx.x * blockDim.x + threadIdx.x;
    if (e >= E) return;
    int s = src[e], d = dst[e];
    int g = batch[s];
    float s0 = assign[2 * s], s1 = assign[2 * s + 1];
    float d0 = assign[2 * d], d1 = assign[2 * d + 1];
    float* a = newadj + 4 * g;
    atomicAdd(a + 0, s0 * d0);
    atomicAdd(a + 1, s0 * d1);
    atomicAdd(a + 2, s1 * d0);
    atomicAdd(a + 3, s1 * d1);
}

__global__ void penalty_kernel(const float* __restrict__ newadj, float* __restrict__ pen, int G) {
    int g = blockIdx.x * blockDim.x + threadIdx.x;
    if (g >= G) return;
    float a00 = newadj[4 * g], a01 = newadj[4 * g + 1];
    float a10 = newadj[4 * g + 2], a11 = newadj[4 * g + 3];
    float r0 = fmaxf(fabsf(a00) + fabsf(a01), EPS_F);
    float r1 = fmaxf(fabsf(a10) + fabsf(a11), EPS_F);
    float d0 = a00 / r0 - 1.0f, d1 = a11 / r1 - 1.0f;
    atomicAdd(pen, (d0 * d0 + d1 * d1) / (2.0f * G));
}

__global__ void sumsq_kernel(const float* __restrict__ p, int n, float* __restrict__ out) {
    __shared__ float sm[256];
    float s = 0.0f;
    for (int i = blockIdx.x * blockDim.x + threadIdx.x; i < n; i += gridDim.x * blockDim.x) {
        float v = p[i]; s += v * v;
    }
    sm[threadIdx.x] = s; __syncthreads();
    for (int o = 128; o > 0; o >>= 1) {
        if (threadIdx.x < o) sm[threadIdx.x] += sm[threadIdx.x + o];
        __syncthreads();
    }
    if (threadIdx.x == 0) atomicAdd(out, sm[0]);
}

__global__ void posnorm_kernel(const float* __restrict__ posraw, const float* __restrict__ ssum,
                               float* __restrict__ pos_out, int n) {
    int i = blockIdx.x * blockDim.x + threadIdx.x;
    if (i >= n) return;
    float p = sqrtf(ssum[0] / (float)n);
    float scale = (p > 1.0f) ? (1.0f / p) : 1.0f;
    pos_out[i] = posraw[i] * scale;
}

// gemb: mean row broadcast to G rows with RMS clip; one block of 512 threads
__global__ void gemb_kernel(const float* __restrict__ colsum, float* __restrict__ gemb_out,
                            int N, int G) {
    __shared__ float sm[HID];
    int f = threadIdx.x;
    float v = colsum[f] / (float)N;
    sm[f] = v * v; __syncthreads();
    for (int o = 256; o > 0; o >>= 1) {
        if (f < o) sm[f] += sm[f + o];
        __syncthreads();
    }
    float p = sqrtf(sm[0] / (float)HID);
    float scale = (p > 1.0f) ? (1.0f / p) : 1.0f;
    float ov = v * scale;
    for (int g = 0; g < G; ++g) gemb_out[(size_t)g * HID + f] = ov;
}

// final head: logp = log_softmax(r @ Wl2 + bl2), one thread per graph
__global__ void head_kernel(const float* __restrict__ r, const float* __restrict__ Wl2,
                            const float* __restrict__ bl2, float* __restrict__ logp, int G) {
    int g = blockIdx.x * blockDim.x + threadIdx.x;
    if (g >= G) return;
    float o[N_CLS];
    #pragma unroll
    for (int c = 0; c < N_CLS; ++c) o[c] = bl2[c];
    const float* rg = r + (size_t)g * HID;
    for (int k = 0; k < HID; ++k) {
        float rv = rg[k];
        #pragma unroll
        for (int c = 0; c < N_CLS; ++c) o[c] += rv * Wl2[k * N_CLS + c];
    }
    float mx = o[0];
    #pragma unroll
    for (int c = 1; c < N_CLS; ++c) mx = fmaxf(mx, o[c]);
    float se = 0.0f;
    #pragma unroll
    for (int c = 0; c < N_CLS; ++c) se += expf(o[c] - mx);
    float lse = mx + logf(se);
    #pragma unroll
    for (int c = 0; c < N_CLS; ++c) logp[g * N_CLS + c] = o[c] - lse;
}

// ---------------- host orchestration ----------------
extern "C" void kernel_launch(void* const* d_in, const int* in_sizes, int n_in,
                              void* d_out, int out_size, void* d_ws, size_t ws_size,
                              hipStream_t stream)
{
    const float* x     = (const float*)d_in[0];
    const int*   ei    = (const int*)d_in[1];
    const int*   src   = ei;
    const int*   dst   = ei + N_EDGES;
    const int*   batch = (const int*)d_in[2];
    const float* W1  = (const float*)d_in[4];   const float* b1  = (const float*)d_in[5];
    const float* W2  = (const float*)d_in[6];   const float* b2  = (const float*)d_in[7];
    const float* W3  = (const float*)d_in[8];   const float* b3  = (const float*)d_in[9];
    const float* Wc1 = (const float*)d_in[10];  const float* bc1 = (const float*)d_in[11];
    const float* Wc2 = (const float*)d_in[12];  const float* bc2 = (const float*)d_in[13];
    const float* Wl1 = (const float*)d_in[14];  const float* bl1 = (const float*)d_in[15];
    const float* Wl2 = (const float*)d_in[16];  const float* bl2 = (const float*)d_in[17];

    // workspace carve-up (256B aligned)
    char* ws = (char*)d_ws;
    size_t o = 0;
    auto alloc = [&](size_t bytes) { size_t r = o; o += (bytes + 255) & ~(size_t)255; return r; };
    __bf16* bufA   = (__bf16*)(ws + alloc((size_t)N_NODES * HID * 2));  // bf16 GEMM out (t / U)
    float*  bufB   = (float*) (ws + alloc((size_t)N_NODES * HID * 4));  // f32 post-agg h
    float*  deg    = (float*) (ws + alloc((size_t)N_NODES * 4));
    float*  dis    = (float*) (ws + alloc((size_t)N_NODES * 4));
    float*  enorm  = (float*) (ws + alloc((size_t)N_EDGES * 4));
    float*  assign = (float*) (ws + alloc((size_t)N_NODES * 2 * 4));
    float*  posraw = (float*) (ws + alloc((size_t)N_GRAPHS * HID * 4));
    float*  rbuf   = (float*) (ws + alloc((size_t)N_GRAPHS * HID * 4));
    float*  colsum = (float*) (ws + alloc((size_t)HID * 4));
    float*  newadj = (float*) (ws + alloc((size_t)N_GRAPHS * 4 * 4));
    float*  scal   = (float*) (ws + alloc(256));
    __bf16* W1t    = (__bf16*)(ws + alloc((size_t)HID * F_IN * 2));
    __bf16* W2t    = (__bf16*)(ws + alloc((size_t)HID * HID * 2));
    __bf16* W3t    = (__bf16*)(ws + alloc((size_t)HID * HID * 2));
    __bf16* Wc1t   = (__bf16*)(ws + alloc((size_t)HID * HID * 2));
    __bf16* Wl1t   = (__bf16*)(ws + alloc((size_t)HID * HID * 2));

    float* out     = (float*)d_out;
    float* logp    = out;                                  // [G, 6]
    float* pos_out = out + (size_t)N_GRAPHS * N_CLS;       // [G, HID]
    float* gemb_out= pos_out + (size_t)N_GRAPHS * HID;     // [G, HID]
    float* pen_out = gemb_out + (size_t)N_GRAPHS * HID;    // [1]

    auto zero = [&](float* p, int n) {
        zero_f32_kernel<<<(n + 255) / 256, 256, 0, stream>>>(p, n);
    };

    // re-zero all accumulators every call (graph replay must be deterministic)
    zero(deg, N_NODES);
    zero(posraw, N_GRAPHS * HID);
    zero(colsum, HID);
    zero(newadj, N_GRAPHS * 4);
    zero(scal, 4);
    zero(pen_out, 1);

    // weight transpose + bf16 convert
    wt_kernel<<<(F_IN * HID + 255) / 256, 256, 0, stream>>>(W1,  W1t,  F_IN, HID);
    wt_kernel<<<(HID * HID + 255) / 256, 256, 0, stream>>>(W2,  W2t,  HID, HID);
    wt_kernel<<<(HID * HID + 255) / 256, 256, 0, stream>>>(W3,  W3t,  HID, HID);
    wt_kernel<<<(HID * HID + 255) / 256, 256, 0, stream>>>(Wc1, Wc1t, HID, HID);
    wt_kernel<<<(HID * HID + 255) / 256, 256, 0, stream>>>(Wl1, Wl1t, HID, HID);

    // degrees / normalization
    deg_kernel<<<(N_EDGES + 255) / 256, 256, 0, stream>>>(dst, deg, N_EDGES);
    dis_kernel<<<(N_NODES + 255) / 256, 256, 0, stream>>>(deg, dis, N_NODES);
    enorm_kernel<<<(N_EDGES + 255) / 256, 256, 0, stream>>>(src, dst, dis, enorm, N_EDGES);

    const int gemmBlocksBig = ((N_NODES / 16) * (HID / 64) + 7) / 8;   // 3200
    const int initBlocks    = (N_NODES * (HID / 4) + 255) / 256;       // 25600
    const int scatBlocks    = (int)(((long long)N_EDGES * (HID / 4) + 255) / 256); // 409600

    // ---- GCN layer 1 ----
    gemm_bf16_kernel<0><<<gemmBlocksBig, 256, 0, stream>>>(x, W1t, nullptr, nullptr, bufA,
                                                           N_NODES, F_IN, HID);
    agg_init_kernel<<<initBlocks, 256, 0, stream>>>(bufA, dis, b1, bufB, N_NODES);
    agg_scatter_kernel<<<scatBlocks, 256, 0, stream>>>(bufA, src, dst, enorm, bufB, N_EDGES);
    // ---- GCN layer 2 ----
    gemm_bf16_kernel<0><<<gemmBlocksBig, 256, 0, stream>>>(bufB, W2t, nullptr, nullptr, bufA,
                                                           N_NODES, HID, HID);
    agg_init_kernel<<<initBlocks, 256, 0, stream>>>(bufA, dis, b2, bufB, N_NODES);
    agg_scatter_kernel<<<scatBlocks, 256, 0, stream>>>(bufA, src, dst, enorm, bufB, N_EDGES);
    // ---- GCN layer 3 ----
    gemm_bf16_kernel<0><<<gemmBlocksBig, 256, 0, stream>>>(bufB, W3t, nullptr, nullptr, bufA,
                                                           N_NODES, HID, HID);
    agg_init_kernel<<<initBlocks, 256, 0, stream>>>(bufA, dis, b3, bufB, N_NODES);
    agg_scatter_kernel<<<scatBlocks, 256, 0, stream>>>(bufA, src, dst, enorm, bufB, N_EDGES);
    // bufB now holds final h (f32)

    // ---- cluster assignment ----
    gemm_bf16_kernel<1><<<gemmBlocksBig, 256, 0, stream>>>(bufB, Wc1t, bc1, nullptr, bufA,
                                                           N_NODES, HID, HID);  // U = tanh(...)
    assign_kernel<<<(N_NODES + 255) / 256, 256, 0, stream>>>(bufA, Wc2, bc2, assign, N_NODES);

    // ---- pooling heads ----
    pos_kernel<<<initBlocks, 256, 0, stream>>>(bufB, assign, batch, posraw, N_NODES);
    colsum_kernel<<<64, HID, 0, stream>>>(bufB, colsum, N_NODES);
    newadj_kernel<<<(N_EDGES + 255) / 256, 256, 0, stream>>>(assign, src, dst, batch,
                                                             newadj, N_EDGES);
    penalty_kernel<<<(N_GRAPHS + 255) / 256, 256, 0, stream>>>(newadj, pen_out, N_GRAPHS);

    // pos RMS clip -> d_out pos section
    sumsq_kernel<<<256, 256, 0, stream>>>(posraw, N_GRAPHS * HID, scal);
    posnorm_kernel<<<(N_GRAPHS * HID + 255) / 256, 256, 0, stream>>>(posraw, scal, pos_out,
                                                                     N_GRAPHS * HID);
    // gemb broadcast + clip -> d_out gemb section
    gemb_kernel<<<1, HID, 0, stream>>>(colsum, gemb_out, N_NODES, N_GRAPHS);

    // ---- classifier MLP on normalized pos ----
    const int gemmBlocksMLP = ((N_GRAPHS / 16) * (HID / 64) + 7) / 8;  // 32
    gemm_bf16_kernel<2><<<gemmBlocksMLP, 256, 0, stream>>>(pos_out, Wl1t, bl1, rbuf, nullptr,
                                                           N_GRAPHS, HID, HID); // relu
    head_kernel<<<(N_GRAPHS + 255) / 256, 256, 0, stream>>>(rbuf, Wl2, bl2, logp, N_GRAPHS);
}
